// Block_82042465288934
// MI455X (gfx1250) — matrix operand order repaired and checked by reading, not compiled, not run on hardware
//
#include <hip/hip_runtime.h>
#include <hip/hip_bf16.h>
#include <math.h>

// ---------------- problem constants ----------------
#define BATCH 8
#define DIM   128
#define HW    4096          // 64*64
#define NPIX  32768         // BATCH*HW
#define C1    32            // DIM/RED
#define N2    392           // K*K*G
#define N2P   400           // padded to 16-multiple
#define WAVES 8             // waves per 256-thread block

typedef __attribute__((ext_vector_type(16))) _Float16 v16h;
typedef __attribute__((ext_vector_type(8)))  _Float16 v8h;
typedef __attribute__((ext_vector_type(4)))  _Float16 v4h;
typedef __attribute__((ext_vector_type(8)))  float    v8f;

// ---------------------------------------------------------------------------
// WMMA fragment loaders (layouts per CDNA5 ISA 7.12.2, wave32)
// A (16x32 f16): lane L holds row m=L&15; elems 0..7 = K[k0 + (L>=16)*8 .. +8)
//                elems 8..15 = K[k0+16 + (L>=16)*8 .. +8)
// ---------------------------------------------------------------------------
__device__ __forceinline__ v16h load_a_frag(const _Float16* __restrict__ A,
                                            int ldA, int lane, int k0) {
  const int m  = lane & 15;
  const int hi = lane >> 4;
  const _Float16* r = A + (size_t)m * ldA + k0 + hi * 8;
  v8h lo = *(const v8h*)(r);
  v8h hh = *(const v8h*)(r + 16);
  v16h a;
#pragma unroll
  for (int e = 0; e < 8; ++e) { a[e] = lo[e]; a[e + 8] = hh[e]; }
  return a;
}

// B (32x16 f16): lane L holds col n=L&15; elems 0..7 = K[k0 + (L>=16)*8 + e], elems 8..15 = K[k0+16 + (L>=16)*8 + e-8]
// Source is pre-packed row-major f16 weight W[n][k] (B[k][n] == W[n][k]):
// two 8-half runs (at +0 and +16) per lane -> two 16-byte loads, zero conversion work.
__device__ __forceinline__ v16h load_b_frag_f16(const _Float16* __restrict__ W,
                                                int ldW, int lane, int n0, int k0) {
  const int n = n0 + (lane & 15);
  const _Float16* r = W + (size_t)n * ldW + k0 + (lane >> 4) * 8;
  v8h lo = *(const v8h*)(r);
  v8h hh = *(const v8h*)(r + 16);
  v16h b;
#pragma unroll
  for (int e = 0; e < 8; ++e) { b[e] = lo[e]; b[e + 8] = hh[e]; }
  return b;
}

// ---------------------------------------------------------------------------
// Kernel W: f32 -> f16 weight pack (zero-pads dst beyond nsrc)
// ---------------------------------------------------------------------------
__global__ void k_f32_to_f16(const float* __restrict__ src, _Float16* __restrict__ dst,
                             int n, int nsrc) {
  const int i = blockIdx.x * 256 + threadIdx.x;
  if (i < n) dst[i] = (i < nsrc) ? (_Float16)src[i] : (_Float16)0.f;
}

// ---------------------------------------------------------------------------
// Kernel 0: NCHW f32 -> NHWC f16 (x16[p][c]) via LDS 32x32 transpose
// ---------------------------------------------------------------------------
__global__ void k_pack_x(const float* __restrict__ x, _Float16* __restrict__ x16) {
  __shared__ float tile[32][33];
  const int b   = blockIdx.z;
  const int c0  = blockIdx.y * 32;
  const int hw0 = blockIdx.x * 32;
  const int tx  = threadIdx.x & 31;
  const int ty  = threadIdx.x >> 5;      // 0..7
#pragma unroll
  for (int r = 0; r < 32; r += 8)
    tile[ty + r][tx] = x[((size_t)(b * DIM + c0 + ty + r)) * HW + hw0 + tx];
  __syncthreads();
#pragma unroll
  for (int r = 0; r < 32; r += 8)
    x16[((size_t)b * HW + hw0 + ty + r) * DIM + c0 + tx] = (_Float16)tile[tx][ty + r];
}

// ---------------------------------------------------------------------------
// Kernel 1: conv1 (1x1, 128->32) + folded BN + ReLU.  M=NPIX N=32 K=128
// One wave: 1 m-tile x 2 n-tiles (full N), A fragment reused across both.
// ---------------------------------------------------------------------------
__global__ void k_conv1(const _Float16* __restrict__ x16,
                        const _Float16* __restrict__ w1h, const float* __restrict__ b1,
                        const float* __restrict__ bn_g, const float* __restrict__ bn_b,
                        const float* __restrict__ bn_mean, const float* __restrict__ bn_var,
                        _Float16* __restrict__ t16) {
  const int lane = threadIdx.x & 31;
  const int mt = blockIdx.x * WAVES + (threadIdx.x >> 5);   // 0..2047
  const _Float16* A = x16 + (size_t)mt * 16 * DIM;
  v8f acc[2] = {};
#pragma unroll
  for (int ks = 0; ks < DIM; ks += 32) {
    v16h a = load_a_frag(A, DIM, lane, ks);
#pragma unroll
    for (int t = 0; t < 2; ++t) {
      v16h b = load_b_frag_f16(w1h, DIM, lane, t * 16, ks);
      acc[t] = __builtin_amdgcn_wmma_f32_16x16x32_f16(false, a, false, b,
                                                      (short)0, acc[t], false, false);
    }
  }
  const int mbase = (lane >> 4) * 8;
#pragma unroll
  for (int t = 0; t < 2; ++t) {
    const int n = t * 16 + (lane & 15);
    const float scale = bn_g[n] * rsqrtf(bn_var[n] + 1e-5f);
    const float beff  = b1[n] * scale + bn_b[n] - bn_mean[n] * scale;
#pragma unroll
    for (int v = 0; v < 8; ++v) {
      float z = acc[t][v] * scale + beff;
      z = z > 0.f ? z : 0.f;
      t16[((size_t)mt * 16 + mbase + v) * C1 + n] = (_Float16)z;
    }
  }
}

// ---------------------------------------------------------------------------
// Kernel 2: conv2 (1x1, 32->392) -> involution weights.  M=NPIX N=400(pad) K=32
// One wave: 1 m-tile x 5 n-tiles (w2h zero-padded to 400 rows, no guards).
// ---------------------------------------------------------------------------
__global__ void k_conv2(const _Float16* __restrict__ t16,
                        const _Float16* __restrict__ w2h, const float* __restrict__ b2,
                        _Float16* __restrict__ wgt16) {
  const int lane = threadIdx.x & 31;
  const int mt = blockIdx.x * WAVES + (threadIdx.x >> 5);   // 0..2047
  const int nt0 = blockIdx.y * 5;                           // 0,5,10,15,20
  const _Float16* A = t16 + (size_t)mt * 16 * C1;
  v8f acc[5] = {};
  v16h a = load_a_frag(A, C1, lane, 0);
#pragma unroll
  for (int t = 0; t < 5; ++t) {
    v16h b = load_b_frag_f16(w2h, C1, lane, (nt0 + t) * 16, 0);
    acc[t] = __builtin_amdgcn_wmma_f32_16x16x32_f16(false, a, false, b,
                                                    (short)0, acc[t], false, false);
  }
  const int mbase = (lane >> 4) * 8;
#pragma unroll
  for (int t = 0; t < 5; ++t) {
    const int n = (nt0 + t) * 16 + (lane & 15);
    const float bias = (n < N2) ? b2[n] : 0.f;
#pragma unroll
    for (int v = 0; v < 8; ++v)
      wgt16[((size_t)mt * 16 + mbase + v) * N2P + n] = (_Float16)(acc[t][v] + bias);
  }
}

// ---------------------------------------------------------------------------
// Kernel 3: involution (7x7, 8 groups of 16 ch) + LayerNorm -> y16 NHWC f16.
// One wave per pixel; lane handles channels 4*lane..4*lane+3 (one group per
// 4 lanes -> weight load is a broadcast; activation taps are 8B contiguous
// per lane, 256B contiguous per wave from the NHWC f16 tensor).
// ---------------------------------------------------------------------------
__global__ void k_invo_ln(const _Float16* __restrict__ x16,
                          const _Float16* __restrict__ wgt16,
                          const float* __restrict__ ln_g, const float* __restrict__ ln_b,
                          _Float16* __restrict__ y16) {
  const int lane = threadIdx.x & 31;
  const int p = blockIdx.x * WAVES + (threadIdx.x >> 5);
  const int b = p >> 12, hw = p & 4095, h = hw >> 6, w = hw & 63;
  const int c0 = lane * 4;
  const int g  = c0 >> 4;
  const _Float16* wg = wgt16 + (size_t)p * N2P + g * 49;
  const _Float16* xb = x16 + (size_t)b * HW * DIM + c0;
  float a0 = 0.f, a1 = 0.f, a2 = 0.f, a3 = 0.f;
#pragma unroll
  for (int i = 0; i < 7; ++i) {
    const int hh = h + i - 3;
    if (hh < 0 || hh >= 64) continue;
#pragma unroll
    for (int j = 0; j < 7; ++j) {
      const int ww = w + j - 3;
      if (ww < 0 || ww >= 64) continue;
      const float wv = (float)wg[i * 7 + j];
      const v4h xv = *(const v4h*)(xb + (size_t)(hh * 64 + ww) * DIM);
      a0 += wv * (float)xv[0];
      a1 += wv * (float)xv[1];
      a2 += wv * (float)xv[2];
      a3 += wv * (float)xv[3];
    }
  }
  float s  = a0 + a1 + a2 + a3;
  float sq = a0 * a0 + a1 * a1 + a2 * a2 + a3 * a3;
#pragma unroll
  for (int m = 16; m >= 1; m >>= 1) {
    s  += __shfl_xor(s,  m, 32);
    sq += __shfl_xor(sq, m, 32);
  }
  const float mean = s * (1.f / 128.f);
  const float var  = sq * (1.f / 128.f) - mean * mean;
  const float rstd = rsqrtf(var + 1e-6f);
  const float vv[4] = {a0, a1, a2, a3};
  v4h o;
#pragma unroll
  for (int t = 0; t < 4; ++t) {
    const int c = c0 + t;
    o[t] = (_Float16)((vv[t] - mean) * rstd * ln_g[c] + ln_b[c]);
  }
  *(v4h*)(y16 + (size_t)p * DIM + c0) = o;
}

// ---------------------------------------------------------------------------
// Kernel 4: pw1 (128->256) + exact GELU.  M=NPIX N=256 K=128
// One wave: 1 m-tile x 4 n-tiles.
// ---------------------------------------------------------------------------
__global__ void k_pw1(const _Float16* __restrict__ y16,
                      const _Float16* __restrict__ wh, const float* __restrict__ bias,
                      _Float16* __restrict__ a16) {
  const int lane = threadIdx.x & 31;
  const int mt = blockIdx.x * WAVES + (threadIdx.x >> 5);   // 0..2047
  const int nt0 = blockIdx.y * 4;                           // 0,4,8,12
  const _Float16* A = y16 + (size_t)mt * 16 * DIM;
  v8f acc[4] = {};
#pragma unroll
  for (int ks = 0; ks < DIM; ks += 32) {
    v16h a = load_a_frag(A, DIM, lane, ks);
#pragma unroll
    for (int t = 0; t < 4; ++t) {
      v16h b = load_b_frag_f16(wh, DIM, lane, (nt0 + t) * 16, ks);
      acc[t] = __builtin_amdgcn_wmma_f32_16x16x32_f16(false, a, false, b,
                                                      (short)0, acc[t], false, false);
    }
  }
  const int mbase = (lane >> 4) * 8;
#pragma unroll
  for (int t = 0; t < 4; ++t) {
    const int n = (nt0 + t) * 16 + (lane & 15);
    const float bn = bias[n];
#pragma unroll
    for (int v = 0; v < 8; ++v) {
      float z = acc[t][v] + bn;
      z = 0.5f * z * (1.f + erff(z * 0.70710678118654752f));   // exact GELU
      a16[((size_t)mt * 16 + mbase + v) * (2 * DIM) + n] = (_Float16)z;
    }
  }
}

// ---------------------------------------------------------------------------
// Kernel 5: pw2 (256->128) + bias + residual, store NCHW f32.  M=NPIX N=128 K=256
// One wave: 1 m-tile x 4 n-tiles.
// ---------------------------------------------------------------------------
__global__ void k_pw2(const _Float16* __restrict__ a16,
                      const _Float16* __restrict__ wh, const float* __restrict__ bias,
                      const float* __restrict__ x, float* __restrict__ out) {
  const int lane = threadIdx.x & 31;
  const int mt = blockIdx.x * WAVES + (threadIdx.x >> 5);   // 0..2047
  const int nt0 = blockIdx.y * 4;                           // 0,4
  const _Float16* A = a16 + (size_t)mt * 16 * (2 * DIM);
  v8f acc[4] = {};
#pragma unroll
  for (int ks = 0; ks < 2 * DIM; ks += 32) {
    v16h a = load_a_frag(A, 2 * DIM, lane, ks);
#pragma unroll
    for (int t = 0; t < 4; ++t) {
      v16h b = load_b_frag_f16(wh, 2 * DIM, lane, (nt0 + t) * 16, ks);
      acc[t] = __builtin_amdgcn_wmma_f32_16x16x32_f16(false, a, false, b,
                                                      (short)0, acc[t], false, false);
    }
  }
  const int mbase = (lane >> 4) * 8;
#pragma unroll
  for (int t = 0; t < 4; ++t) {
    const int c = (nt0 + t) * 16 + (lane & 15);
    const float bn = bias[c];
#pragma unroll
    for (int v = 0; v < 8; ++v) {
      const int p = mt * 16 + mbase + v;
      const int b = p >> 12, hw = p & 4095;
      const size_t gi = ((size_t)b * DIM + c) * HW + hw;
      out[gi] = acc[t][v] + bn + x[gi];
    }
  }
}

// ---------------------------------------------------------------------------
extern "C" void kernel_launch(void* const* d_in, const int* in_sizes, int n_in,
                              void* d_out, int out_size, void* d_ws, size_t ws_size,
                              hipStream_t stream) {
  const float* x       = (const float*)d_in[0];
  const float* conv1_w = (const float*)d_in[1];
  const float* conv1_b = (const float*)d_in[2];
  const float* bn_g    = (const float*)d_in[3];
  const float* bn_b    = (const float*)d_in[4];
  const float* bn_mean = (const float*)d_in[5];
  const float* bn_var  = (const float*)d_in[6];
  const float* conv2_w = (const float*)d_in[7];
  const float* conv2_b = (const float*)d_in[8];
  const float* ln_g    = (const float*)d_in[9];
  const float* ln_b    = (const float*)d_in[10];
  const float* pw1_w   = (const float*)d_in[11];
  const float* pw1_b   = (const float*)d_in[12];
  const float* pw2_w   = (const float*)d_in[13];
  const float* pw2_b   = (const float*)d_in[14];
  float* out = (float*)d_out;

  // Workspace layout (bytes, 256-aligned): all L2-resident (~62 MB of 192 MB)
  char* ws = (char*)d_ws;
  _Float16* x16   = (_Float16*)(ws + 0);          // NPIX*128 f16  =  8.4 MB
  _Float16* t16   = (_Float16*)(ws + 8388608);    // NPIX*32  f16  =  2.1 MB
  _Float16* wgt16 = (_Float16*)(ws + 10485760);   // NPIX*400 f16  = 26.2 MB
  _Float16* y16   = (_Float16*)(ws + 36700160);   // NPIX*128 f16  =  8.4 MB
  _Float16* a16   = (_Float16*)(ws + 45088768);   // NPIX*256 f16  = 16.8 MB
  _Float16* w1h   = (_Float16*)(ws + 61865984);   // 32*128        =  8 KB
  _Float16* w2h   = (_Float16*)(ws + 61874176);   // 400*32 (pad)  = 25 KB
  _Float16* pw1h  = (_Float16*)(ws + 61899264);   // 256*128       = 64 KB
  _Float16* pw2h  = (_Float16*)(ws + 61964800);   // 128*256       = 64 KB

  // Pack weights to f16 once (w2 zero-padded from 392 to 400 rows).
  k_f32_to_f16<<<dim3((32 * 128 + 255) / 256),  256, 0, stream>>>(conv1_w, w1h, 32 * 128, 32 * 128);
  k_f32_to_f16<<<dim3((N2P * C1 + 255) / 256),  256, 0, stream>>>(conv2_w, w2h, N2P * C1, N2 * C1);
  k_f32_to_f16<<<dim3((256 * 128 + 255) / 256), 256, 0, stream>>>(pw1_w, pw1h, 256 * 128, 256 * 128);
  k_f32_to_f16<<<dim3((128 * 256 + 255) / 256), 256, 0, stream>>>(pw2_w, pw2h, 128 * 256, 128 * 256);

  k_pack_x <<<dim3(HW / 32, DIM / 32, BATCH), 256, 0, stream>>>(x, x16);
  k_conv1  <<<dim3(NPIX / 16 / WAVES, 1), 256, 0, stream>>>(x16, w1h, conv1_b,
                                                            bn_g, bn_b, bn_mean, bn_var, t16);
  k_conv2  <<<dim3(NPIX / 16 / WAVES, 5), 256, 0, stream>>>(t16, w2h, conv2_b, wgt16);
  k_invo_ln<<<dim3(NPIX / WAVES),         256, 0, stream>>>(x16, wgt16, ln_g, ln_b, y16);
  k_pw1    <<<dim3(NPIX / 16 / WAVES, 4), 256, 0, stream>>>(y16, pw1h, pw1_b, a16);
  k_pw2    <<<dim3(NPIX / 16 / WAVES, 2), 256, 0, stream>>>(a16, pw2h, pw2_b, x, out);
}
